// GRUStage2_54589034332756
// MI455X (gfx1250) — compile-verified, matrix operand order checked by reading, not run on hardware
//
#include <hip/hip_runtime.h>
#include <stdint.h>

typedef __attribute__((ext_vector_type(16))) _Float16 v16h;
typedef __attribute__((ext_vector_type(8)))  _Float16 v8h;
typedef __attribute__((ext_vector_type(8)))  float    v8f;

// ---------------------------------------------------------------------------
// elementwise f32 -> f16
// ---------------------------------------------------------------------------
__global__ void k_f32_to_f16(const float* __restrict__ s, _Float16* __restrict__ d, int n)
{
  for (int i = blockIdx.x * blockDim.x + threadIdx.x; i < n; i += gridDim.x * blockDim.x)
    d[i] = (_Float16)s[i];
}

// ---------------------------------------------------------------------------
// tiled transpose: src f32 [R,C] -> dst f16 [C,R]
// ---------------------------------------------------------------------------
__global__ void k_transpose_to_f16(const float* __restrict__ src, _Float16* __restrict__ dst,
                                   int R, int C)
{
  __shared__ float tile[32][33];
  int tx = threadIdx.x, ty = threadIdx.y;
  int c = blockIdx.x * 32 + tx;
  int r = blockIdx.y * 32 + ty;
  if (r < R && c < C) tile[ty][tx] = src[(size_t)r * C + c];
  __syncthreads();
  int dr = blockIdx.x * 32 + ty;   // dst row = src col
  int dc = blockIdx.y * 32 + tx;   // dst col = src row
  if (dr < C && dc < R) dst[(size_t)dr * R + dc] = (_Float16)tile[tx][ty];
}

// ---------------------------------------------------------------------------
// WMMA GEMM: C[M,N] f32 = A[M,K]f16 * W[N,K]f16^T (+bias[N]) (+addsrc[M,N])
// One wave computes a 16x64 output strip (4 accumulators, A-frag reused 4x).
// A-frag per lane (m = lane%16, h = lane/16): a[0..7]=A[m][kb+8h..], a[8..15]=A[m][kb+16+8h..]
// ---------------------------------------------------------------------------
__global__ void __launch_bounds__(256)
k_wmma_gemm(const _Float16* __restrict__ A, const _Float16* __restrict__ W,
            const float* __restrict__ bias, const float* __restrict__ addsrc,
            float* __restrict__ C, int M, int N, int K)
{
  const int wavesPerBlk = blockDim.x >> 5;
  int wid = blockIdx.x * wavesPerBlk + (threadIdx.x >> 5);
  const int tilesN = N >> 6;               // 64-wide N strips
  const int total  = (M >> 4) * tilesN;
  if (wid >= total) return;                // wave-uniform: EXEC stays all-ones
  const int tm   = wid / tilesN;
  const int tn   = wid - tm * tilesN;
  const int lane = threadIdx.x & 31;
  const int hsel = lane >> 4;
  const int mrow = lane & 15;

  const _Float16* arow  = A + (size_t)(tm * 16 + mrow) * K + hsel * 8;
  const _Float16* brow0 = W + (size_t)(tn * 64 +  0 + mrow) * K + hsel * 8;
  const _Float16* brow1 = W + (size_t)(tn * 64 + 16 + mrow) * K + hsel * 8;
  const _Float16* brow2 = W + (size_t)(tn * 64 + 32 + mrow) * K + hsel * 8;
  const _Float16* brow3 = W + (size_t)(tn * 64 + 48 + mrow) * K + hsel * 8;

  v8f acc0 = {}, acc1 = {}, acc2 = {}, acc3 = {};
  for (int k0 = 0; k0 < K; k0 += 32) {
    v8h al = *(const v8h*)(arow + k0);
    v8h ah = *(const v8h*)(arow + k0 + 16);
    v16h a = __builtin_shufflevector(al, ah, 0,1,2,3,4,5,6,7,8,9,10,11,12,13,14,15);
#define BFRAG(p) __builtin_shufflevector(*(const v8h*)((p) + k0), *(const v8h*)((p) + k0 + 16), \
                                         0,1,2,3,4,5,6,7,8,9,10,11,12,13,14,15)
    v16h b0 = BFRAG(brow0);
    v16h b1 = BFRAG(brow1);
    v16h b2 = BFRAG(brow2);
    v16h b3 = BFRAG(brow3);
#undef BFRAG
    acc0 = __builtin_amdgcn_wmma_f32_16x16x32_f16(false, a, false, b0, (short)0, acc0, false, false);
    acc1 = __builtin_amdgcn_wmma_f32_16x16x32_f16(false, a, false, b1, (short)0, acc1, false, false);
    acc2 = __builtin_amdgcn_wmma_f32_16x16x32_f16(false, a, false, b2, (short)0, acc2, false, false);
    acc3 = __builtin_amdgcn_wmma_f32_16x16x32_f16(false, a, false, b3, (short)0, acc3, false, false);
    __builtin_prefetch(arow + k0 + 256, 0, 0);   // global_prefetch_b8
  }

#define EPI(ACC, NB) {                                                     \
    int ncol = tn * 64 + (NB) * 16 + mrow;                                 \
    float bv = bias ? bias[ncol] : 0.f;                                    \
    _Pragma("unroll")                                                      \
    for (int r = 0; r < 8; ++r) {                                          \
      size_t idx = (size_t)(tm * 16 + hsel * 8 + r) * N + ncol;            \
      float v = ACC[r] + bv;                                               \
      if (addsrc) v += addsrc[idx];                                        \
      C[idx] = v;                                                          \
    } }
  EPI(acc0, 0) EPI(acc1, 1) EPI(acc2, 2) EPI(acc3, 3)
#undef EPI
}

// ---------------------------------------------------------------------------
// row-wise LayerNorm (biased var, eps=1e-5): one block per row
// ---------------------------------------------------------------------------
__global__ void __launch_bounds__(256)
k_ln_rows(const float* __restrict__ src, const float* __restrict__ g,
          const float* __restrict__ b, float* __restrict__ dst, int C)
{
  __shared__ float red[256];
  int row = blockIdx.x, tid = threadIdx.x;
  const float* x = src + (size_t)row * C;
  float s = 0.f;
  for (int i = tid; i < C; i += 256) s += x[i];
  red[tid] = s; __syncthreads();
  for (int st = 128; st > 0; st >>= 1) { if (tid < st) red[tid] += red[tid + st]; __syncthreads(); }
  float mean = red[0] / C; __syncthreads();
  float v = 0.f;
  for (int i = tid; i < C; i += 256) { float d = x[i] - mean; v += d * d; }
  red[tid] = v; __syncthreads();
  for (int st = 128; st > 0; st >>= 1) { if (tid < st) red[tid] += red[tid + st]; __syncthreads(); }
  float rstd = rsqrtf(red[0] / C + 1e-5f);
  float* o = dst + (size_t)row * C;
  for (int i = tid; i < C; i += 256) o[i] = (x[i] - mean) * rstd * g[i] + b[i];
}

// ---------------------------------------------------------------------------
// Encoder GRU scan: grid of 2 blocks (block 0 = forward, block 1 = backward),
// 512 threads; gi (x@Wih^T+bih) precomputed via WMMA; whh pre-transposed f16
// [512 rows k][1536 cols j] so lane access is coalesced. h lives in LDS.
// out is [T,1024] interleaved (fwd cols 0..511, bwd cols 512..1023).
// ---------------------------------------------------------------------------
__global__ void __launch_bounds__(512)
k_gru_scan(const float* __restrict__ giF, const float* __restrict__ giB,
           const _Float16* __restrict__ wF, const _Float16* __restrict__ wB,
           const float* __restrict__ bF, const float* __restrict__ bB,
           float* __restrict__ out, float* __restrict__ hT, int T)
{
  const bool rev = (blockIdx.x == 1);
  const float*    gi  = rev ? giB : giF;
  const _Float16* wt  = rev ? wB  : wF;
  const float*    bhh = rev ? bB  : bF;
  const int outoff = rev ? 512 : 0;
  __shared__ float hs[512];
  int j = threadIdx.x;
  hs[j] = 0.f;
  __syncthreads();
  const float b_r = bhh[j], b_z = bhh[512 + j], b_n = bhh[1024 + j];
  for (int step = 0; step < T; ++step) {
    int t = rev ? (T - 1 - step) : step;
    float ar = 0.f, az = 0.f, an = 0.f;
    for (int k = 0; k < 512; ++k) {
      float hk = hs[k];
      const _Float16* wr = wt + (size_t)k * 1536;
      ar += (float)wr[j]        * hk;
      az += (float)wr[512 + j]  * hk;
      an += (float)wr[1024 + j] * hk;
    }
    const float* git = gi + (size_t)t * 1536;
    float r = 1.f / (1.f + __expf(-(git[j]        + ar + b_r)));
    float z = 1.f / (1.f + __expf(-(git[512 + j]  + az + b_z)));
    float n = tanhf(git[1024 + j] + r * (an + b_n));
    float hp = hs[j];
    float hn = (1.f - z) * n + z * hp;
    __syncthreads();
    hs[j] = hn;
    out[(size_t)t * 1024 + outoff + j] = hn;
    __syncthreads();
  }
  hT[(rev ? 512 : 0) + j] = hs[j];
}

// ---------------------------------------------------------------------------
// hidden = LN(hidden_cat) @ fc_enc.w^T + fc_enc.b ; hidden_cat[l] = [hf_l|hb_l]
// ---------------------------------------------------------------------------
__global__ void __launch_bounds__(512)
k_enc_hidden(const float* __restrict__ hT, const float* __restrict__ g,
             const float* __restrict__ b, const float* __restrict__ fw,
             const float* __restrict__ fb, float* __restrict__ hidden)
{
  __shared__ float row[1024];
  __shared__ float red[512];
  int tid = threadIdx.x;
  for (int l = 0; l < 2; ++l) {
    float a0 = hT[l * 1024 + tid], a1 = hT[l * 1024 + 512 + tid];
    red[tid] = a0 + a1; __syncthreads();
    for (int s = 256; s > 0; s >>= 1) { if (tid < s) red[tid] += red[tid + s]; __syncthreads(); }
    float mean = red[0] * (1.f / 1024.f); __syncthreads();
    float d0 = a0 - mean, d1 = a1 - mean;
    red[tid] = d0 * d0 + d1 * d1; __syncthreads();
    for (int s = 256; s > 0; s >>= 1) { if (tid < s) red[tid] += red[tid + s]; __syncthreads(); }
    float rstd = rsqrtf(red[0] * (1.f / 1024.f) + 1e-5f); __syncthreads();
    row[tid]       = d0 * rstd * g[tid]       + b[tid];
    row[512 + tid] = d1 * rstd * g[512 + tid] + b[512 + tid];
    __syncthreads();
    float acc = fb[tid];
    const float* wr = fw + (size_t)tid * 1024;
    for (int k = 0; k < 1024; ++k) acc += wr[k] * row[k];
    hidden[l * 512 + tid] = acc;
    __syncthreads();
  }
}

// ---------------------------------------------------------------------------
// Decoder: single persistent block, 1024 threads, 512 teacher-forced steps.
// keyT f16 [512,4096] (k-major), val16 f16 [4096,512], all weights k-major f16.
// ---------------------------------------------------------------------------
__global__ void __launch_bounds__(1024)
k_decoder(const int* __restrict__ y, const float* __restrict__ embd,
          const _Float16* __restrict__ wqT, const float* __restrict__ bq,
          const _Float16* __restrict__ keyT, const _Float16* __restrict__ val16,
          const _Float16* __restrict__ wih0T, const _Float16* __restrict__ whh0T,
          const float* __restrict__ bih0, const float* __restrict__ bhh0,
          const _Float16* __restrict__ wih1T, const _Float16* __restrict__ whh1T,
          const float* __restrict__ bih1, const float* __restrict__ bhh1,
          const float* __restrict__ lnh_g, const float* __restrict__ lnh_b,
          const float* __restrict__ lno_g, const float* __restrict__ lno_b,
          const _Float16* __restrict__ fc1T, const float* __restrict__ b1,
          const _Float16* __restrict__ fc2T, const float* __restrict__ b2,
          const float* __restrict__ hidden0, float* __restrict__ out)
{
  __shared__ float h0[512], h1[512], h0r[512], h1r[512];
  __shared__ float q[512], gin[1024];
  __shared__ float gg[1536], gh[1536];
  __shared__ float pre[1536];
  __shared__ float p1[256];
  __shared__ float scores[4096];
  __shared__ float red[1024];
  int tid = threadIdx.x;
  if (tid < 512) { h0[tid] = hidden0[tid]; h1[tid] = hidden0[512 + tid]; }
  __syncthreads();

  for (int step = 0; step < 512; ++step) {
    int tok = (step == 0) ? 0 : y[step - 1];
    // ---- q = Wq @ s + bq  (s = h1) ; gin[0:512] = emb[tok] ----
    if (tid < 512) {
      float acc = bq[tid];
      for (int k = 0; k < 512; ++k) acc += (float)wqT[(size_t)k * 512 + tid] * h1[k];
      q[tid] = acc;
      gin[tid] = embd[(size_t)tok * 512 + tid];
    }
    __syncthreads();
    // ---- attention scores + softmax over T=4096 ----
    float lmax = -3.4e38f;
    for (int t = tid; t < 4096; t += 1024) {
      float acc = 0.f;
      for (int k = 0; k < 512; ++k) acc += (float)keyT[(size_t)k * 4096 + t] * q[k];
      scores[t] = acc;
      lmax = fmaxf(lmax, acc);
    }
    red[tid] = lmax; __syncthreads();
    for (int s = 512; s > 0; s >>= 1) { if (tid < s) red[tid] = fmaxf(red[tid], red[tid + s]); __syncthreads(); }
    float mx = red[0]; __syncthreads();
    float lsum = 0.f;
    for (int t = tid; t < 4096; t += 1024) { float e = __expf(scores[t] - mx); scores[t] = e; lsum += e; }
    red[tid] = lsum; __syncthreads();
    for (int s = 512; s > 0; s >>= 1) { if (tid < s) red[tid] += red[tid + s]; __syncthreads(); }
    float inv = 1.f / red[0]; __syncthreads();
    // ---- context: gin[512:1024] = alphas @ val + s ----
    if (tid < 512) {
      float acc = 0.f;
      for (int t = 0; t < 4096; ++t) acc += scores[t] * (float)val16[(size_t)t * 512 + tid];
      gin[512 + tid] = acc * inv + h1[tid];
    }
    __syncthreads();
    // ---- GRU layer 0 (in=gin[1024], h=h0) ----
    for (int i = tid; i < 1536; i += 1024) {
      float ai = bih0[i];
      for (int k = 0; k < 1024; ++k) ai += (float)wih0T[(size_t)k * 1536 + i] * gin[k];
      float ah = bhh0[i];
      for (int k = 0; k < 512; ++k) ah += (float)whh0T[(size_t)k * 1536 + i] * h0[k];
      gg[i] = ai; gh[i] = ah;
    }
    __syncthreads();
    if (tid < 512) {
      float r = 1.f / (1.f + __expf(-(gg[tid] + gh[tid])));
      float z = 1.f / (1.f + __expf(-(gg[512 + tid] + gh[512 + tid])));
      float n = tanhf(gg[1024 + tid] + r * gh[1024 + tid]);
      h0r[tid] = (1.f - z) * n + z * h0[tid];
    }
    __syncthreads();
    // ---- GRU layer 1 (in=h0r raw, h=h1) ----
    for (int i = tid; i < 1536; i += 1024) {
      float ai = bih1[i];
      for (int k = 0; k < 512; ++k) ai += (float)wih1T[(size_t)k * 1536 + i] * h0r[k];
      float ah = bhh1[i];
      for (int k = 0; k < 512; ++k) ah += (float)whh1T[(size_t)k * 1536 + i] * h1[k];
      gg[i] = ai; gh[i] = ah;
    }
    __syncthreads();
    if (tid < 512) {
      float r = 1.f / (1.f + __expf(-(gg[tid] + gh[tid])));
      float z = 1.f / (1.f + __expf(-(gg[512 + tid] + gh[512 + tid])));
      float n = tanhf(gg[1024 + tid] + r * gh[1024 + tid]);
      h1r[tid] = (1.f - z) * n + z * h1[tid];
    }
    __syncthreads();
    // ---- next-step state: h = LN(raw h) with ln_hid_dec ----
    {
      float v0 = (tid < 512) ? h0r[tid] : 0.f;
      red[tid] = v0; __syncthreads();
      for (int s = 512; s > 0; s >>= 1) { if (tid < s) red[tid] += red[tid + s]; __syncthreads(); }
      float mean = red[0] * (1.f / 512.f); __syncthreads();
      float d = v0 - mean;
      red[tid] = (tid < 512) ? d * d : 0.f; __syncthreads();
      for (int s = 512; s > 0; s >>= 1) { if (tid < s) red[tid] += red[tid + s]; __syncthreads(); }
      float rstd = rsqrtf(red[0] * (1.f / 512.f) + 1e-5f); __syncthreads();
      if (tid < 512) h0[tid] = d * rstd * lnh_g[tid] + lnh_b[tid];
    }
    {
      float v0 = (tid < 512) ? h1r[tid] : 0.f;
      red[tid] = v0; __syncthreads();
      for (int s = 512; s > 0; s >>= 1) { if (tid < s) red[tid] += red[tid + s]; __syncthreads(); }
      float mean = red[0] * (1.f / 512.f); __syncthreads();
      float d = v0 - mean;
      red[tid] = (tid < 512) ? d * d : 0.f; __syncthreads();
      for (int s = 512; s > 0; s >>= 1) { if (tid < s) red[tid] += red[tid + s]; __syncthreads(); }
      float rstd = rsqrtf(red[0] * (1.f / 512.f) + 1e-5f); __syncthreads();
      if (tid < 512) h1[tid] = d * rstd * lnh_g[tid] + lnh_b[tid];
    }
    // ---- pre = LN([h1r | gin], 1536) with ln_out_dec ----
    {
      float x0 = (tid < 512) ? h1r[tid] : gin[tid - 512];
      float x1 = (tid < 512) ? gin[512 + tid] : 0.f;
      red[tid] = x0 + x1; __syncthreads();
      for (int s = 512; s > 0; s >>= 1) { if (tid < s) red[tid] += red[tid + s]; __syncthreads(); }
      float mean = red[0] * (1.f / 1536.f); __syncthreads();
      float d0 = x0 - mean, d1 = x1 - mean;
      red[tid] = d0 * d0 + ((tid < 512) ? d1 * d1 : 0.f); __syncthreads();
      for (int s = 512; s > 0; s >>= 1) { if (tid < s) red[tid] += red[tid + s]; __syncthreads(); }
      float rstd = rsqrtf(red[0] * (1.f / 1536.f) + 1e-5f); __syncthreads();
      pre[tid] = d0 * rstd * lno_g[tid] + lno_b[tid];
      if (tid < 512) pre[1024 + tid] = d1 * rstd * lno_g[1024 + tid] + lno_b[1024 + tid];
    }
    __syncthreads();
    // ---- fc1 + tanh-GELU ----
    if (tid < 256) {
      float acc = b1[tid];
      for (int k = 0; k < 1536; ++k) acc += (float)fc1T[(size_t)k * 256 + tid] * pre[k];
      float g3 = acc * acc * acc;
      p1[tid] = 0.5f * acc * (1.f + tanhf(0.7978845608028654f * (acc + 0.044715f * g3)));
    }
    __syncthreads();
    // ---- fc2 -> logits ----
    if (tid < 32) {
      float acc = b2[tid];
      for (int k = 0; k < 256; ++k) acc += (float)fc2T[(size_t)k * 32 + tid] * p1[k];
      out[(size_t)step * 32 + tid] = acc;
    }
    __syncthreads();
  }
}

// ---------------------------------------------------------------------------
extern "C" void kernel_launch(void* const* d_in, const int* in_sizes, int n_in,
                              void* d_out, int out_size, void* d_ws, size_t ws_size,
                              hipStream_t stream)
{
  (void)in_sizes; (void)n_in; (void)out_size; (void)ws_size;
  const float* x      = (const float*)d_in[0];
  const int*   y      = (const int*)d_in[1];
  const float *wih0f = (const float*)d_in[2],  *whh0f = (const float*)d_in[3],
              *bih0f = (const float*)d_in[4],  *bhh0f = (const float*)d_in[5];
  const float *wih0b = (const float*)d_in[6],  *whh0b = (const float*)d_in[7],
              *bih0b = (const float*)d_in[8],  *bhh0b = (const float*)d_in[9];
  const float *wih1f = (const float*)d_in[10], *whh1f = (const float*)d_in[11],
              *bih1f = (const float*)d_in[12], *bhh1f = (const float*)d_in[13];
  const float *wih1b = (const float*)d_in[14], *whh1b = (const float*)d_in[15],
              *bih1b = (const float*)d_in[16], *bhh1b = (const float*)d_in[17];
  const float *lnhe_g = (const float*)d_in[18], *lnhe_b = (const float*)d_in[19];
  const float *lne_g  = (const float*)d_in[20], *lne_b  = (const float*)d_in[21];
  const float *fce_w  = (const float*)d_in[22], *fce_b  = (const float*)d_in[23];
  const float *skip_w = (const float*)d_in[24], *skip_b = (const float*)d_in[25];
  const float *embd   = (const float*)d_in[26];
  const float *wq_w   = (const float*)d_in[27], *wq_b = (const float*)d_in[28];
  const float *wk_w   = (const float*)d_in[29], *wk_b = (const float*)d_in[30];
  const float *wv_w   = (const float*)d_in[31], *wv_b = (const float*)d_in[32];
  const float *dwih0 = (const float*)d_in[33], *dwhh0 = (const float*)d_in[34],
              *dbih0 = (const float*)d_in[35], *dbhh0 = (const float*)d_in[36];
  const float *dwih1 = (const float*)d_in[37], *dwhh1 = (const float*)d_in[38],
              *dbih1 = (const float*)d_in[39], *dbhh1 = (const float*)d_in[40];
  const float *lnhd_g = (const float*)d_in[41], *lnhd_b = (const float*)d_in[42];
  const float *lnod_g = (const float*)d_in[43], *lnod_b = (const float*)d_in[44];
  const float *fc1_w  = (const float*)d_in[45], *fc1_b  = (const float*)d_in[46];
  const float *fc2_w  = (const float*)d_in[47], *fc2_b  = (const float*)d_in[48];

  char* ws = (char*)d_ws;
  size_t off = 0;
  auto alloc = [&](size_t bytes) -> char* {
    char* p = ws + off; off += (bytes + 255) & ~(size_t)255; return p;
  };

  _Float16* X16    = (_Float16*)alloc(4096ull * 768 * 2);
  float*    GI_F   = (float*)alloc(4096ull * 1536 * 4);
  char*     GI_Braw= alloc(4096ull * 1536 * 4);
  float*    GI_B   = (float*)GI_Braw;
  float*    SEQA   = (float*)alloc(4096ull * 1024 * 4);
  float*    SEQB   = (float*)alloc(4096ull * 1024 * 4);
  _Float16* SEQA16 = (_Float16*)alloc(4096ull * 1024 * 2);
  _Float16* W16_WIH0F = (_Float16*)alloc(1536ull * 768 * 2);
  _Float16* W16_WIH0B = (_Float16*)alloc(1536ull * 768 * 2);
  _Float16* W16_WIH1F = (_Float16*)alloc(1536ull * 1024 * 2);
  _Float16* W16_WIH1B = (_Float16*)alloc(1536ull * 1024 * 2);
  _Float16* W16_SKIP  = (_Float16*)alloc(1024ull * 768 * 2);
  _Float16* W16_WK    = (_Float16*)alloc(512ull * 1024 * 2);
  _Float16* W16_WV    = (_Float16*)alloc(512ull * 1024 * 2);
  _Float16* WT_WHH0F  = (_Float16*)alloc(512ull * 1536 * 2);
  _Float16* WT_WHH0B  = (_Float16*)alloc(512ull * 1536 * 2);
  _Float16* WT_WHH1F  = (_Float16*)alloc(512ull * 1536 * 2);
  _Float16* WT_WHH1B  = (_Float16*)alloc(512ull * 1536 * 2);
  _Float16* WT_WQ     = (_Float16*)alloc(512ull * 512 * 2);
  _Float16* WT_DWIH0  = (_Float16*)alloc(1024ull * 1536 * 2);
  _Float16* WT_DWHH0  = (_Float16*)alloc(512ull * 1536 * 2);
  _Float16* WT_DWIH1  = (_Float16*)alloc(512ull * 1536 * 2);
  _Float16* WT_DWHH1  = (_Float16*)alloc(512ull * 1536 * 2);
  _Float16* WT_FC1    = (_Float16*)alloc(1536ull * 256 * 2);
  _Float16* WT_FC2    = (_Float16*)alloc(256ull * 32 * 2);
  float*    HT        = (float*)alloc(4ull * 512 * 4);
  float*    HIDDEN    = (float*)alloc(2ull * 512 * 4);
  // overlays (buffers freed after encoder scans complete)
  float*    ENC_OUT = GI_F;                               // [4096,1024] f32
  _Float16* ENC16   = (_Float16*)GI_Braw;                 // [4096,1024] f16 (8MB)
  float*    KEYT    = (float*)(GI_Braw + 8388608);        // [4096,512]  f32 (8MB)
  float*    VALT    = (float*)(GI_Braw + 16777216);       // [4096,512]  f32 (8MB)
  _Float16* KEYTT16 = SEQA16;                             // [512,4096]  f16 (4MB)
  _Float16* VAL16   = SEQA16 + 2097152;                   // [4096,512]  f16 (4MB)

  dim3 tb(32, 32);
  auto cvt = [&](const float* s, _Float16* d, int n) {
    k_f32_to_f16<<<2048, 256, 0, stream>>>(s, d, n);
  };
  auto tr = [&](const float* s, _Float16* d, int R, int C) {
    k_transpose_to_f16<<<dim3((C + 31) / 32, (R + 31) / 32), tb, 0, stream>>>(s, d, R, C);
  };
  auto gemm = [&](const _Float16* A, const _Float16* W, const float* bias,
                  const float* add, float* Cc, int M, int N, int K) {
    int tiles = (M / 16) * (N / 64);
    k_wmma_gemm<<<(tiles + 7) / 8, 256, 0, stream>>>(A, W, bias, add, Cc, M, N, K);
  };

  // ---- precompute f16 operands ----
  cvt(x, X16, 4096 * 768);
  cvt(wih0f, W16_WIH0F, 1536 * 768);
  cvt(wih0b, W16_WIH0B, 1536 * 768);
  cvt(wih1f, W16_WIH1F, 1536 * 1024);
  cvt(wih1b, W16_WIH1B, 1536 * 1024);
  cvt(skip_w, W16_SKIP, 1024 * 768);
  cvt(wk_w, W16_WK, 512 * 1024);
  cvt(wv_w, W16_WV, 512 * 1024);
  tr(whh0f, WT_WHH0F, 1536, 512);
  tr(whh0b, WT_WHH0B, 1536, 512);
  tr(whh1f, WT_WHH1F, 1536, 512);
  tr(whh1b, WT_WHH1B, 1536, 512);
  tr(wq_w,  WT_WQ,    512, 512);
  tr(dwih0, WT_DWIH0, 1536, 1024);
  tr(dwhh0, WT_DWHH0, 1536, 512);
  tr(dwih1, WT_DWIH1, 1536, 512);
  tr(dwhh1, WT_DWHH1, 1536, 512);
  tr(fc1_w, WT_FC1,   256, 1536);
  tr(fc2_w, WT_FC2,   32, 256);

  // ---- encoder layer 0 ----
  gemm(X16, W16_WIH0F, bih0f, nullptr, GI_F, 4096, 1536, 768);
  gemm(X16, W16_WIH0B, bih0b, nullptr, GI_B, 4096, 1536, 768);
  k_gru_scan<<<2, 512, 0, stream>>>(GI_F, GI_B, WT_WHH0F, WT_WHH0B, bhh0f, bhh0b, SEQA, HT, 4096);
  cvt(SEQA, SEQA16, 4096 * 1024);
  // ---- encoder layer 1 ----
  gemm(SEQA16, W16_WIH1F, bih1f, nullptr, GI_F, 4096, 1536, 1024);
  gemm(SEQA16, W16_WIH1B, bih1b, nullptr, GI_B, 4096, 1536, 1024);
  k_gru_scan<<<2, 512, 0, stream>>>(GI_F, GI_B, WT_WHH1F, WT_WHH1B, bhh1f, bhh1b, SEQB, HT + 1024, 4096);
  // ---- enc_out = x@skip^T + b + LN(inp2) ----
  k_ln_rows<<<4096, 256, 0, stream>>>(SEQB, lne_g, lne_b, SEQA, 1024);
  gemm(X16, W16_SKIP, skip_b, SEQA, ENC_OUT, 4096, 1024, 768);
  cvt(ENC_OUT, ENC16, 4096 * 1024);
  // ---- key/value projections ----
  gemm(ENC16, W16_WK, wk_b, nullptr, KEYT, 4096, 512, 1024);
  gemm(ENC16, W16_WV, wv_b, nullptr, VALT, 4096, 512, 1024);
  tr(KEYT, KEYTT16, 4096, 512);
  cvt(VALT, VAL16, 4096 * 512);
  // ---- initial decoder hidden ----
  k_enc_hidden<<<1, 512, 0, stream>>>(HT, lnhe_g, lnhe_b, fce_w, fce_b, HIDDEN);
  // ---- decoder ----
  k_decoder<<<1, 1024, 0, stream>>>(y, embd, WT_WQ, wq_b, KEYTT16, VAL16,
                                    WT_DWIH0, WT_DWHH0, dbih0, dbhh0,
                                    WT_DWIH1, WT_DWHH1, dbih1, dbhh1,
                                    lnhd_g, lnhd_b, lnod_g, lnod_b,
                                    WT_FC1, fc1_b, WT_FC2, fc2_b,
                                    HIDDEN, (float*)d_out);
}
